// Model_86311662780422
// MI455X (gfx1250) — compile-verified, matrix operand order checked by reading, not run on hardware
//
#include <hip/hip_runtime.h>
#include <hip/hip_bf16.h>

// ---------------------------------------------------------------------------
// RGCN conv + MLP + graph readout + cosine margin loss for MI455X (gfx1250).
//
// Strategy: all matmuls are 64x64 with fp32 inputs; weights fit in L2, so the
// work is gather-bound. We downconvert to bf16 and drive
// v_wmma_f32_16x16x32_bf16 (fp32 accumulate): 16-row tile x K=64 x N=64 is
// 8 WMMA ops per tile. edge_type is sorted -> 16-edge tiles are relation-
// uniform except at <=31 boundaries (endpoint check, rare VALU fallback).
// Scatter aggregation uses global_atomic_add_f32 (unsafeAtomicAdd).
// ---------------------------------------------------------------------------

#define N_NODES  131072
#define N_EDGES  524288
#define N_GRAPHS 16384
#define EMB      64
#define NREL     32

typedef __attribute__((ext_vector_type(16))) __bf16 v16bf;
typedef __attribute__((ext_vector_type(8)))  __bf16 v8bf;
typedef __attribute__((ext_vector_type(8)))  float  v8f;

__device__ __forceinline__ v8f wmma_bf16(v16bf a, v16bf b, v8f c) {
  return __builtin_amdgcn_wmma_f32_16x16x32_bf16(
      /*neg_a=*/false, a, /*neg_b=*/false, b,
      /*c_mod=*/(short)0, c, /*reuse_a=*/false, /*reuse_b=*/false);
}

// Fragment storage layout for a 64x64 (KxN) weight matrix as WMMA B operands:
//   frag f = ktile*4 + ntile  (ktile 0..1 over K=64, ntile 0..3 over N=64)
//   element addr = matrix_base + f*512 + lane*16 + e   (bf16 units)
//   mapping: N = ntile*16 + (lane&15); khalf = lane>>4
//            K = ktile*32 + 8*khalf + (e<8 ? e : e+8)
// Per-lane fragment = 16 contiguous bf16 = one aligned 32B load.
#define FRAG_MAT_ELEMS 4096   // 8 frags * 32 lanes * 16 elems

// ---------------------------------------------------------------------------
// Kernel 0: convert W_rel (32 mats), W_root, W1, W2 to bf16 fragment layout.
// ---------------------------------------------------------------------------
__global__ __launch_bounds__(256) void k_cvt_weights(
    const float* __restrict__ Wrel, const float* __restrict__ Wroot,
    const float* __restrict__ W1,   const float* __restrict__ W2,
    unsigned short* __restrict__ Wb_raw)
{
  const int idx = blockIdx.x * 256 + threadIdx.x;   // 35 * 4096 total
  if (idx >= 35 * FRAG_MAT_ELEMS) return;
  const int m    = idx >> 12;
  const int f    = (idx >> 9) & 7;
  const int lane = (idx >> 4) & 31;
  const int e    = idx & 15;
  const int ktile = f >> 2, ntile = f & 3;
  const int col   = ntile * 16 + (lane & 15);
  const int khalf = lane >> 4;
  const int k     = ktile * 32 + 8 * khalf + ((e < 8) ? e : e + 8);
  float v;
  if (m < 32)       v = Wrel[((size_t)m * 64 + k) * 64 + col];
  else if (m == 32) v = Wroot[k * 64 + col];
  else if (m == 33) v = W1[k * 64 + col];
  else              v = W2[k * 64 + col];
  ((__bf16*)Wb_raw)[idx] = (__bf16)v;
}

// Build the two A-matrix fragments (16 rows x K=64) for one lane from a
// contiguous fp32 row pointer. A layout (ISA 7.12.2, 16-bit A 16x32):
//   lanes 0-15: M=lane,   K = {0..7, 16..23}+ktile*32
//   lanes16-31: M=lane-16,K = {8..15,24..31}+ktile*32
__device__ __forceinline__ void build_a_frags(const float* __restrict__ xr,
                                              int khalf, v16bf& a0, v16bf& a1) {
  const int o = 8 * khalf;
#pragma unroll
  for (int j = 0; j < 8; ++j) {
    a0[j]     = (__bf16)xr[o + j];
    a0[j + 8] = (__bf16)xr[o + 16 + j];
    a1[j]     = (__bf16)xr[32 + o + j];
    a1[j + 8] = (__bf16)xr[32 + o + 16 + j];
  }
}

// ---------------------------------------------------------------------------
// Kernel 1: per-edge messages msg = x[src] @ W_rel[type], mean-aggregated
// into agg[dst] via fp32 atomics; degree counts into cnt.
// One wave per 16-edge tile; 8 waves per block.
// ---------------------------------------------------------------------------
__global__ __launch_bounds__(256) void k_edge_msg(
    const int* __restrict__ node_ids, const int* __restrict__ edge_src,
    const int* __restrict__ edge_dst, const int* __restrict__ edge_type,
    const float* __restrict__ emb,    const float* __restrict__ Wrel_f32,
    const unsigned short* __restrict__ Wb_raw,
    float* __restrict__ agg, float* __restrict__ cnt)
{
  const __bf16* Wb = (const __bf16*)Wb_raw;
  const int lane  = threadIdx.x & 31;
  const int wave  = threadIdx.x >> 5;
  const int tile  = blockIdx.x * 8 + wave;
  const int base  = tile * 16;

  if (lane < 16) unsafeAtomicAdd(&cnt[edge_dst[base + lane]], 1.0f);

  const int r0 = edge_type[base];
  const int r1 = edge_type[base + 15];

  if (r0 == r1) {               // wave-uniform branch: EXEC all-ones for WMMA
    const int row   = lane & 15;
    const int khalf = lane >> 4;
    const int nid   = node_ids[edge_src[base + row]];
    v16bf a0, a1;
    build_a_frags(emb + (size_t)nid * 64, khalf, a0, a1);

    const __bf16* wbase = Wb + (size_t)r0 * FRAG_MAT_ELEMS + lane * 16;
    v8f c[4];
#pragma unroll
    for (int nt = 0; nt < 4; ++nt) {
      v8f acc = {};
      acc = wmma_bf16(a0, *(const v16bf*)(wbase + ((0 * 4 + nt) << 9)), acc);
      acc = wmma_bf16(a1, *(const v16bf*)(wbase + ((1 * 4 + nt) << 9)), acc);
      c[nt] = acc;
    }
    // C layout: element (g, lane) -> row M = g + 8*khalf, col = (lane&15)+nt*16
    int dsts[8];
#pragma unroll
    for (int g = 0; g < 8; ++g) dsts[g] = edge_dst[base + g + 8 * khalf];
#pragma unroll
    for (int nt = 0; nt < 4; ++nt) {
      const int col = nt * 16 + (lane & 15);
#pragma unroll
      for (int g = 0; g < 8; ++g)
        unsafeAtomicAdd(&agg[(size_t)dsts[g] * 64 + col], c[nt][g]);
    }
  } else {                      // rare (<=31/32768 tiles): relation boundary
    const int col0 = lane * 2;
    for (int m = 0; m < 16; ++m) {
      const int e   = base + m;
      const int r   = edge_type[e];
      const int nid = node_ids[edge_src[e]];
      const float* xr = emb + (size_t)nid * 64;
      const float* wr = Wrel_f32 + (size_t)r * 4096;
      float s0 = 0.f, s1 = 0.f;
#pragma unroll 8
      for (int k = 0; k < 64; ++k) {
        const float xv = xr[k];
        s0 = fmaf(xv, wr[k * 64 + col0],     s0);
        s1 = fmaf(xv, wr[k * 64 + col0 + 1], s1);
      }
      const int d = edge_dst[e];
      unsafeAtomicAdd(&agg[(size_t)d * 64 + col0],     s0);
      unsafeAtomicAdd(&agg[(size_t)d * 64 + col0 + 1], s1);
    }
  }
}

// Reload A fragments from a row-major bf16 16x64 LDS tile.
__device__ __forceinline__ void load_a_from_lds(const __bf16* __restrict__ buf,
                                                int row, int khalf,
                                                v16bf& a0, v16bf& a1) {
  const __bf16* rp = buf + row * 64;
  const int o = 8 * khalf;
  v8bf lo0 = *(const v8bf*)(rp + o);
  v8bf hi0 = *(const v8bf*)(rp + o + 16);
  v8bf lo1 = *(const v8bf*)(rp + 32 + o);
  v8bf hi1 = *(const v8bf*)(rp + 48 + o);
  a0 = __builtin_shufflevector(lo0, hi0, 0,1,2,3,4,5,6,7,8,9,10,11,12,13,14,15);
  a1 = __builtin_shufflevector(lo1, hi1, 0,1,2,3,4,5,6,7,8,9,10,11,12,13,14,15);
}

// ---------------------------------------------------------------------------
// Kernel 2: h = relu(agg/max(cnt,1) + x@W_root + cb); h = relu(h@W1+b1);
//           h_out = h@W2 + b2.  One wave per 16-node tile; C->A transpose
//           between chained matmuls goes through a per-wave LDS tile.
// ---------------------------------------------------------------------------
__global__ __launch_bounds__(256) void k_node_mlp(
    const int* __restrict__ node_ids, const float* __restrict__ emb,
    const unsigned short* __restrict__ Wb_raw,
    const float* __restrict__ agg, const float* __restrict__ cnt,
    const float* __restrict__ conv_bias, const float* __restrict__ b1,
    const float* __restrict__ b2, float* __restrict__ h_out)
{
  __shared__ __align__(32) __bf16 lds[8][16 * 64];   // 2KB per wave
  const __bf16* Wb = (const __bf16*)Wb_raw;
  const int lane    = threadIdx.x & 31;
  const int wave    = threadIdx.x >> 5;
  const int base    = (blockIdx.x * 8 + wave) * 16;
  const int row     = lane & 15;
  const int khalf   = lane >> 4;
  const int colbase = lane & 15;
  __bf16* buf = lds[wave];

  const __bf16* wroot = Wb + 32 * FRAG_MAT_ELEMS + lane * 16;
  const __bf16* w1    = Wb + 33 * FRAG_MAT_ELEMS + lane * 16;
  const __bf16* w2    = Wb + 34 * FRAG_MAT_ELEMS + lane * 16;

  // ---- conv: x @ W_root ----
  const int nid = node_ids[base + row];
  v16bf a0, a1;
  build_a_frags(emb + (size_t)nid * 64, khalf, a0, a1);
  v8f c[4];
#pragma unroll
  for (int nt = 0; nt < 4; ++nt) {
    v8f acc = {};
    acc = wmma_bf16(a0, *(const v16bf*)(wroot + ((0 * 4 + nt) << 9)), acc);
    acc = wmma_bf16(a1, *(const v16bf*)(wroot + ((1 * 4 + nt) << 9)), acc);
    c[nt] = acc;
  }
  // + agg/max(cnt,1) + conv_bias, relu; stage bf16 tile to LDS (row-major)
#pragma unroll
  for (int nt = 0; nt < 4; ++nt) {
    const int col = nt * 16 + colbase;
    const float bias = conv_bias[col];
#pragma unroll
    for (int g = 0; g < 8; ++g) {
      const int nrow = base + g + 8 * khalf;
      const float cv = fmaxf(cnt[nrow], 1.0f);
      float v = c[nt][g] + agg[(size_t)nrow * 64 + col] / cv + bias;
      buf[(g + 8 * khalf) * 64 + col] = (__bf16)fmaxf(v, 0.f);
    }
  }
  __syncthreads();

  // ---- linear 1: relu(h @ W1 + b1) ----
  load_a_from_lds(buf, row, khalf, a0, a1);
#pragma unroll
  for (int nt = 0; nt < 4; ++nt) {
    v8f acc = {};
    acc = wmma_bf16(a0, *(const v16bf*)(w1 + ((0 * 4 + nt) << 9)), acc);
    acc = wmma_bf16(a1, *(const v16bf*)(w1 + ((1 * 4 + nt) << 9)), acc);
    c[nt] = acc;
  }
  __syncthreads();   // tile reads done in all waves before overwrite
#pragma unroll
  for (int nt = 0; nt < 4; ++nt) {
    const int col = nt * 16 + colbase;
    const float bias = b1[col];
#pragma unroll
    for (int g = 0; g < 8; ++g)
      buf[(g + 8 * khalf) * 64 + col] = (__bf16)fmaxf(c[nt][g] + bias, 0.f);
  }
  __syncthreads();

  // ---- linear 2: h @ W2 + b2 -> h_out ----
  load_a_from_lds(buf, row, khalf, a0, a1);
#pragma unroll
  for (int nt = 0; nt < 4; ++nt) {
    v8f acc = {};
    acc = wmma_bf16(a0, *(const v16bf*)(w2 + ((0 * 4 + nt) << 9)), acc);
    acc = wmma_bf16(a1, *(const v16bf*)(w2 + ((1 * 4 + nt) << 9)), acc);
    c[nt] = acc;
  }
#pragma unroll
  for (int nt = 0; nt < 4; ++nt) {
    const int col = nt * 16 + colbase;
    const float bias = b2[col];
#pragma unroll
    for (int g = 0; g < 8; ++g)
      h_out[(size_t)(base + g + 8 * khalf) * 64 + col] = c[nt][g] + bias;
  }
}

// ---------------------------------------------------------------------------
// Kernel 3: q[batch_id[n]] += h[n]  (fp32 atomics; one thread per 2 columns)
// ---------------------------------------------------------------------------
__global__ __launch_bounds__(256) void k_segsum(
    const float* __restrict__ h, const int* __restrict__ batch_id,
    float* __restrict__ q)
{
  const int idx  = blockIdx.x * 256 + threadIdx.x;   // N_NODES * 32
  const int node = idx >> 5;
  const int cp   = (idx & 31) * 2;
  const float2 hv = *(const float2*)(h + (size_t)node * 64 + cp);
  const int b = batch_id[node];
  unsafeAtomicAdd(&q[(size_t)b * 64 + cp],     hv.x);
  unsafeAtomicAdd(&q[(size_t)b * 64 + cp + 1], hv.y);
}

// ---------------------------------------------------------------------------
// Kernel 4: cosine scores + margin loss. One wave per graph.
// out[0:G]=loss, out[G:2G]=golden, out[2G:3G]=corrupt.
// ---------------------------------------------------------------------------
__global__ __launch_bounds__(256) void k_score(
    const float* __restrict__ q, const float* __restrict__ emb,
    const int* __restrict__ answers, const int* __restrict__ corrupted,
    float* __restrict__ out)
{
  const int g    = blockIdx.x * 8 + (threadIdx.x >> 5);
  const int lane = threadIdx.x & 31;
  const float2 qv = *(const float2*)(q + (size_t)g * 64 + lane * 2);
  const int ai = answers[g], ci = corrupted[g];
  const float2 av = *(const float2*)(emb + (size_t)ai * 64 + lane * 2);
  const float2 cv = *(const float2*)(emb + (size_t)ci * 64 + lane * 2);
  float qa = qv.x * av.x + qv.y * av.y;
  float qc = qv.x * cv.x + qv.y * cv.y;
  float qq = qv.x * qv.x + qv.y * qv.y;
  float aa = av.x * av.x + av.y * av.y;
  float cc = cv.x * cv.x + cv.y * cv.y;
#pragma unroll
  for (int m = 16; m > 0; m >>= 1) {
    qa += __shfl_xor(qa, m);
    qc += __shfl_xor(qc, m);
    qq += __shfl_xor(qq, m);
    aa += __shfl_xor(aa, m);
    cc += __shfl_xor(cc, m);
  }
  if (lane == 0) {
    const float golden  = qa / sqrtf(qq * aa);
    const float corrupt = qc / sqrtf(qq * cc);
    out[g]                = fmaxf(0.f, 1.0f - golden + corrupt);
    out[N_GRAPHS + g]     = golden;
    out[2 * N_GRAPHS + g] = corrupt;
  }
}

// ---------------------------------------------------------------------------
// Workspace layout (bytes, 256-aligned)
// ---------------------------------------------------------------------------
#define OFF_AGG 0ull                                    // N_NODES*64*4 = 32 MiB
#define OFF_CNT (OFF_AGG + (size_t)N_NODES * 64 * 4)    // N_NODES*4
#define OFF_H   (OFF_CNT + (size_t)N_NODES * 4)         // N_NODES*64*4
#define OFF_Q   (OFF_H   + (size_t)N_NODES * 64 * 4)    // N_GRAPHS*64*4
#define OFF_WB  (OFF_Q   + (size_t)N_GRAPHS * 64 * 4)   // 35*4096*2

extern "C" void kernel_launch(void* const* d_in, const int* in_sizes, int n_in,
                              void* d_out, int out_size, void* d_ws, size_t ws_size,
                              hipStream_t stream) {
  const int*   node_ids  = (const int*)d_in[0];
  const int*   edge_src  = (const int*)d_in[1];
  const int*   edge_dst  = (const int*)d_in[2];
  const int*   edge_type = (const int*)d_in[3];
  const int*   batch_id  = (const int*)d_in[4];
  const int*   answers   = (const int*)d_in[5];
  const int*   corrupted = (const int*)d_in[6];
  const float* emb       = (const float*)d_in[7];
  const float* Wrel      = (const float*)d_in[8];
  const float* Wroot     = (const float*)d_in[9];
  const float* conv_bias = (const float*)d_in[10];
  const float* W1        = (const float*)d_in[11];
  const float* b1        = (const float*)d_in[12];
  const float* W2        = (const float*)d_in[13];
  const float* b2        = (const float*)d_in[14];

  char* wsb = (char*)d_ws;
  float*          agg = (float*)(wsb + OFF_AGG);
  float*          cnt = (float*)(wsb + OFF_CNT);
  float*          h   = (float*)(wsb + OFF_H);
  float*          q   = (float*)(wsb + OFF_Q);
  unsigned short* Wb  = (unsigned short*)(wsb + OFF_WB);
  float* out = (float*)d_out;

  // zero accumulators (capture-safe async memsets)
  hipMemsetAsync(wsb + OFF_AGG, 0, (size_t)N_NODES * 64 * 4 + (size_t)N_NODES * 4, stream);
  hipMemsetAsync(wsb + OFF_Q,   0, (size_t)N_GRAPHS * 64 * 4, stream);

  k_cvt_weights<<<(35 * FRAG_MAT_ELEMS) / 256, 256, 0, stream>>>(Wrel, Wroot, W1, W2, Wb);
  k_edge_msg<<<N_EDGES / (16 * 8), 256, 0, stream>>>(node_ids, edge_src, edge_dst,
                                                     edge_type, emb, Wrel, Wb, agg, cnt);
  k_node_mlp<<<N_NODES / (16 * 8), 256, 0, stream>>>(node_ids, emb, Wb, agg, cnt,
                                                     conv_bias, b1, b2, h);
  k_segsum<<<(N_NODES * 32) / 256, 256, 0, stream>>>(h, batch_id, q);
  k_score<<<N_GRAPHS / 8, 256, 0, stream>>>(q, emb, answers, corrupted, out);
}
